// GATPolicy_89249420411244
// MI455X (gfx1250) — matrix operand order repaired and checked by reading, not compiled, not run on hardware
//
#include <hip/hip_runtime.h>

#define NN   10000
#define EE   320000
#define ET   (EE + NN)      // edges + self loops = 330000
#define HH   8
#define CC   64
#define HCC  512
#define DIN  64
#define ACTD 12
#define GG   64
#define SLOPE 0.2f
#define EPSV  1e-16f
#define KB   64             // K-halves staged in LDS per TDM step

typedef _Float16 half_t;
typedef __attribute__((ext_vector_type(16))) _Float16 v16h;
typedef __attribute__((ext_vector_type(8)))  _Float16 v8h;
typedef __attribute__((ext_vector_type(8)))  float    v8f;

// TDM descriptor groups (6-arg builtin form on this toolchain)
typedef __attribute__((ext_vector_type(4))) unsigned int tdm_u4;
typedef __attribute__((ext_vector_type(8))) int          tdm_i8;
typedef __attribute__((ext_vector_type(4))) int          tdm_i4;

// ---------- monotone float<->uint key for atomic segment-max ----------
__device__ __forceinline__ unsigned fkey(float f) {
  unsigned u = __float_as_uint(f);
  return (u >> 31) ? ~u : (u | 0x80000000u);
}
__device__ __forceinline__ float fkey_inv(unsigned k) {
  return __uint_as_float((k >> 31) ? (k ^ 0x80000000u) : ~k);
}

// ---------- TDM 2-D tile load: global (row stride = stride0 elems) -> LDS ----------
// Descriptor bit-packing per CDNA5 ISA §8.3 (group 0) / §8.4 (group 1).
// data_size = 2 bytes (f16). Groups 2/3 (and trailing group) zero => 2-D tensor.
__device__ __forceinline__ void tdm_load_2d(unsigned ldsAddr, const half_t* gptr,
                                            unsigned tensor_d0, unsigned tensor_d1,
                                            unsigned tile_d0, unsigned tile_d1,
                                            unsigned long long stride0) {
  unsigned long long ga = (unsigned long long)(const void*)gptr;
  tdm_u4 g0;
  g0[0] = 1u;                                            // count=1, user mode
  g0[1] = ldsAddr;                                       // lds_addr [63:32]
  g0[2] = (unsigned)ga;                                  // global_addr[31:0]
  g0[3] = (unsigned)((ga >> 32) & 0x01FFFFFFu)           // global_addr[56:32]
        | 0x80000000u;                                   // type=2 ("image") [127:126]
  tdm_i8 g1;
  g1[0] = (int)(1u << 16);                               // data_size=1 (2 bytes)
  g1[1] = (int)((tensor_d0 & 0xFFFFu) << 16);            // tensor_dim0[15:0] @ [63:48]
  g1[2] = (int)((tensor_d0 >> 16) | ((tensor_d1 & 0xFFFFu) << 16)); // dim0 hi | dim1 lo
  g1[3] = (int)((tensor_d1 >> 16) | (tile_d0 << 16));    // dim1 hi | tile_dim0
  g1[4] = (int)(tile_d1 & 0xFFFFu);                      // tile_dim1 (tile_dim2=0)
  g1[5] = (int)(unsigned)(stride0 & 0xFFFFFFFFu);        // tensor_dim0_stride[31:0]
  g1[6] = (int)(unsigned)((stride0 >> 32) & 0xFFFFu);    // stride hi (dim1_stride lo = 0)
  g1[7] = 0;
  tdm_i4 g2 = {0, 0, 0, 0};
  tdm_i4 g3 = {0, 0, 0, 0};
  tdm_i8 g4 = {0, 0, 0, 0, 0, 0, 0, 0};
  __builtin_amdgcn_tensor_load_to_lds(g0, g1, g2, g3, g4, 0);
}

// ---------- small conversion / setup kernels ----------
__global__ void k_cvt_f16(const float* __restrict__ in, half_t* __restrict__ out, int n) {
  int t = blockIdx.x * blockDim.x + threadIdx.x;
  if (t < n) out[t] = (half_t)in[t];
}

// W [K, 512] f32  ->  WT [512, K] f16 (transposed so tile rows are contiguous in K)
__global__ void k_cvt_wT(const float* __restrict__ W, half_t* __restrict__ WT, int K) {
  int t = blockIdx.x * blockDim.x + threadIdx.x;
  if (t < K * HCC) {
    int k = t / HCC, n = t % HCC;
    WT[(size_t)n * K + k] = (half_t)W[t];
  }
}

// coef[h] = sum_c We[h*64+c] * a_e[h*64+c]   (edge proj collapses to scalar per head)
__global__ void k_coef(const float* __restrict__ We, const float* __restrict__ ae,
                       float* __restrict__ coef) {
  int h = threadIdx.x;
  if (h < HH) {
    float s = 0.f;
    #pragma unroll 8
    for (int c = 0; c < CC; ++c) s += We[h * CC + c] * ae[h * CC + c];
    coef[h] = s;
  }
}

// self-loop attr = mean of incoming edge attrs
__global__ void k_loop_accum(const int* __restrict__ ei, const float* __restrict__ eattr,
                             float* __restrict__ deg, float* __restrict__ esum) {
  int e = blockIdx.x * blockDim.x + threadIdx.x;
  if (e < EE) {
    int dst = ei[EE + e];
    atomicAdd(&deg[dst], 1.0f);
    atomicAdd(&esum[dst], eattr[e]);
  }
}
__global__ void k_loop_final(const float* __restrict__ deg, const float* __restrict__ esum,
                             float* __restrict__ lattr) {
  int n = blockIdx.x * blockDim.x + threadIdx.x;
  if (n < NN) lattr[n] = esum[n] / fmaxf(deg[n], 1.0f);
}

// ---------- WMMA GEMM with TDM->LDS staging ----------
// C[10000,512] = A[10000,K](f16) @ B[K,512]; B supplied transposed BT[512,K].
// Block (256 thr = 8 waves) computes a 16(M) x 128(N) strip; wave w owns N-tile w.
// Per K-step of 64: wave 0 TDM-loads A tile (16x64) and BT tile (128x64) into LDS,
// s_wait_tensorcnt 0, barrier, then each wave runs 2 x v_wmma_f32_16x16x32_f16
// from ds_load_b128 fragments (ISA 16-bit A/B lane layouts).
__global__ void k_gemm_wmma(const half_t* __restrict__ A, const half_t* __restrict__ BT,
                            float* __restrict__ Cout, int K) {
  __shared__ __align__(16) half_t sA[16 * KB];    // 2 KB
  __shared__ __align__(16) half_t sB[128 * KB];   // 16 KB
  const int wave   = threadIdx.x >> 5;
  const int lane   = threadIdx.x & 31;
  const int bm     = blockIdx.x >> 2;             // 0..624  (M strip)
  const int bn     = blockIdx.x & 3;              // 0..3    (N super-tile)
  const int laneHi = lane >> 4;
  const int l16    = lane & 15;

  const unsigned ldsA = (unsigned)(size_t)(void*)sA;
  const unsigned ldsB = (unsigned)(size_t)(void*)sB;
  const half_t* gA = A  + (size_t)(bm * 16) * K;
  const half_t* gB = BT + (size_t)(bn * 128) * K;

  v8f acc = {};
  for (int kb = 0; kb < K; kb += KB) {
    if (threadIdx.x < 32) {                       // wave 0 drives the TDM
      tdm_load_2d(ldsA, gA + kb, (unsigned)K, 16u,  KB, 16u,  (unsigned long long)K);
      tdm_load_2d(ldsB, gB + kb, (unsigned)K, 128u, KB, 128u, (unsigned long long)K);
      __builtin_amdgcn_s_wait_tensorcnt(0);
    }
    __syncthreads();                              // LDS tiles ready
    #pragma unroll
    for (int k2 = 0; k2 < KB; k2 += 32) {
      const v8h* pa = (const v8h*)(sA + l16 * KB + k2 + laneHi * 8);
      const v8h* pb = (const v8h*)(sB + (wave * 16 + l16) * KB + k2 + laneHi * 8);
      v8h a0 = pa[0], a1 = pa[2];                 // K offsets +0 / +16 halves
      v8h b0 = pb[0], b1 = pb[2];                 // K offsets +0 / +16 halves
      v16h av, bv;
      #pragma unroll
      for (int j = 0; j < 8; ++j) {
        av[j] = a0[j]; av[8 + j] = a1[j];
        bv[j] = b0[j]; bv[8 + j] = b1[j];
      }
      acc = __builtin_amdgcn_wmma_f32_16x16x32_f16(false, av, false, bv,
                                                   (short)0, acc, false, false);
    }
    __syncthreads();                              // all reads done before next TDM write
  }
  // C/D layout: VGPR r -> row (r + laneHi*8), col l16
  float* out = Cout + (size_t)(bm * 16 + laneHi * 8) * HCC + bn * 128 + wave * 16 + l16;
  #pragma unroll
  for (int r = 0; r < 8; ++r) out[(size_t)r * HCC] = acc[r];
}

// s_src[n,h], s_dst[n,h] = <h[n,h,:], a_src/a_dst[h,:]>
__global__ void k_scores(const float* __restrict__ hproj, const float* __restrict__ as,
                         const float* __restrict__ ad, float* __restrict__ ssrc,
                         float* __restrict__ sdst) {
  int t = blockIdx.x * blockDim.x + threadIdx.x;
  if (t < NN * HH) {
    int n = t >> 3, h = t & 7;
    const float* hp = hproj + (size_t)n * HCC + h * CC;
    float s0 = 0.f, s1 = 0.f;
    #pragma unroll 8
    for (int c = 0; c < CC; ++c) { s0 += hp[c] * as[h * CC + c]; s1 += hp[c] * ad[h * CC + c]; }
    ssrc[t] = s0; sdst[t] = s1;
  }
}

// pass A: alpha = leaky(s_src[src]+s_dst[dst]+attr*coef); atomic segment-max per (dst,h)
__global__ void k_edge_alpha(const int* __restrict__ ei, const float* __restrict__ eattr,
                             const float* __restrict__ lattr, const float* __restrict__ ssrc,
                             const float* __restrict__ sdst, const float* __restrict__ coef,
                             float* __restrict__ alpha, unsigned* __restrict__ amax) {
  int e = blockIdx.x * blockDim.x + threadIdx.x;
  if (e >= ET) return;
  int src, dst; float a;
  if (e < EE) { src = ei[e]; dst = ei[EE + e]; a = eattr[e]; }
  else        { src = dst = e - EE;            a = lattr[src]; }
  #pragma unroll
  for (int h = 0; h < HH; ++h) {
    float al = ssrc[src * HH + h] + sdst[dst * HH + h] + a * coef[h];
    al = (al > 0.f) ? al : SLOPE * al;
    alpha[(size_t)e * HH + h] = al;
    atomicMax(&amax[dst * HH + h], fkey(al));
  }
}

// pass B: ex = exp(alpha - amax[dst]); atomic segment-sum denom
__global__ void k_edge_ex(const int* __restrict__ ei, float* __restrict__ alpha,
                          const unsigned* __restrict__ amax, float* __restrict__ denom) {
  int e = blockIdx.x * blockDim.x + threadIdx.x;
  if (e >= ET) return;
  int dst = (e < EE) ? ei[EE + e] : (e - EE);
  #pragma unroll
  for (int h = 0; h < HH; ++h) {
    float ex = __expf(alpha[(size_t)e * HH + h] - fkey_inv(amax[dst * HH + h]));
    alpha[(size_t)e * HH + h] = ex;
    atomicAdd(&denom[dst * HH + h], ex);
  }
}

// pass C: hagg[dst] += h[src] * ex/(denom[dst]+eps); one thread per (edge, head, 4-chan)
__global__ void k_edge_agg(const int* __restrict__ ei, const float* __restrict__ exv,
                           const float* __restrict__ denom, const float* __restrict__ hproj,
                           float* __restrict__ hagg) {
  long long t = (long long)blockIdx.x * blockDim.x + threadIdx.x;
  if (t >= (long long)ET * (HH * 16)) return;
  int e  = (int)(t >> 7);
  int r  = (int)(t & 127);
  int h  = r >> 4;
  int c  = (r & 15) * 4;
  int src, dst;
  if (e < EE) { src = ei[e]; dst = ei[EE + e]; }
  else        { src = dst = e - EE; }
  float w = exv[(size_t)e * HH + h] / (denom[dst * HH + h] + EPSV);
  const float4 v = *(const float4*)(hproj + (size_t)src * HCC + h * CC + c);
  float* o = hagg + (size_t)dst * HCC + h * CC + c;
  atomicAdd(o + 0, v.x * w);
  atomicAdd(o + 1, v.y * w);
  atomicAdd(o + 2, v.z * w);
  atomicAdd(o + 3, v.w * w);
}

// bias + relu -> f16 activation for next layer / head
__global__ void k_finalize(const float* __restrict__ hagg, const float* __restrict__ b,
                           half_t* __restrict__ h16) {
  int t = blockIdx.x * blockDim.x + threadIdx.x;
  if (t < NN * HCC) {
    float v = hagg[t] + b[t & (HCC - 1)];
    h16[t] = (half_t)fmaxf(v, 0.f);
  }
}

// ---------- head: tanh(h @ Wl + bl), mean pool per graph ----------
__global__ void k_cnt(const int* __restrict__ batch, float* __restrict__ cnt) {
  int n = blockIdx.x * blockDim.x + threadIdx.x;
  if (n < NN) atomicAdd(&cnt[batch[n]], 1.0f);
}
__global__ void k_head(const half_t* __restrict__ h16, const float* __restrict__ Wl,
                       const float* __restrict__ bl, const int* __restrict__ batch,
                       float* __restrict__ pooled) {
  int t = blockIdx.x * blockDim.x + threadIdx.x;
  if (t >= NN * ACTD) return;
  int n = t / ACTD, d = t % ACTD;
  const half_t* hp = h16 + (size_t)n * HCC;
  float s = bl[d];
  #pragma unroll 8
  for (int k = 0; k < HCC; ++k) s += (float)hp[k] * Wl[k * ACTD + d];
  atomicAdd(&pooled[batch[n] * ACTD + d], tanhf(s));
}
__global__ void k_pool_div(float* __restrict__ pooled, const float* __restrict__ cnt) {
  int t = blockIdx.x * blockDim.x + threadIdx.x;
  if (t < GG * ACTD) pooled[t] /= fmaxf(cnt[t / ACTD], 1.0f);
}

// -------------------------------------------------------------------
extern "C" void kernel_launch(void* const* d_in, const int* in_sizes, int n_in,
                              void* d_out, int out_size, void* d_ws, size_t ws_size,
                              hipStream_t stream) {
  (void)in_sizes; (void)n_in; (void)out_size; (void)ws_size;

  const float* x     = (const float*)d_in[0];
  const int*   ei    = (const int*)d_in[1];
  const float* eattr = (const float*)d_in[2];
  const int*   batch = (const int*)d_in[3];
  const float* W[3]  = {(const float*)d_in[4],  (const float*)d_in[10], (const float*)d_in[16]};
  const float* We[3] = {(const float*)d_in[5],  (const float*)d_in[11], (const float*)d_in[17]};
  const float* As[3] = {(const float*)d_in[6],  (const float*)d_in[12], (const float*)d_in[18]};
  const float* Ad[3] = {(const float*)d_in[7],  (const float*)d_in[13], (const float*)d_in[19]};
  const float* Ae[3] = {(const float*)d_in[8],  (const float*)d_in[14], (const float*)d_in[20]};
  const float* B[3]  = {(const float*)d_in[9],  (const float*)d_in[15], (const float*)d_in[21]};
  const float* Wl    = (const float*)d_in[22];
  const float* bl    = (const float*)d_in[23];
  float* pooled      = (float*)d_out;

  // workspace bump allocator (256B aligned regions)
  char* wp = (char*)d_ws;
  auto alloc = [&](size_t bytes) -> void* {
    void* p = (void*)wp;
    wp += (bytes + 255) & ~(size_t)255;
    return p;
  };
  half_t*   x16   = (half_t*)  alloc((size_t)NN * DIN * 2);
  half_t*   h16   = (half_t*)  alloc((size_t)NN * HCC * 2);
  half_t*   WT16  = (half_t*)  alloc((size_t)HCC * HCC * 2);
  float*    hproj = (float*)   alloc((size_t)NN * HCC * 4);
  float*    hagg  = (float*)   alloc((size_t)NN * HCC * 4);
  float*    ssrc  = (float*)   alloc((size_t)NN * HH * 4);
  float*    sdst  = (float*)   alloc((size_t)NN * HH * 4);
  unsigned* amax  = (unsigned*)alloc((size_t)NN * HH * 4);
  float*    denom = (float*)   alloc((size_t)NN * HH * 4);
  float*    alpha = (float*)   alloc((size_t)ET * HH * 4);
  float*    lattr = (float*)   alloc((size_t)NN * 4);
  float*    deg   = (float*)   alloc((size_t)NN * 4);
  float*    esum  = (float*)   alloc((size_t)NN * 4);
  float*    coef  = (float*)   alloc(256);
  float*    cnt   = (float*)   alloc(256);

  const int T = 256;
  // self-loop attrs (fill_value='mean')
  (void)hipMemsetAsync(deg,  0, (size_t)NN * 4, stream);
  (void)hipMemsetAsync(esum, 0, (size_t)NN * 4, stream);
  k_loop_accum<<<(EE + T - 1) / T, T, 0, stream>>>(ei, eattr, deg, esum);
  k_loop_final<<<(NN + T - 1) / T, T, 0, stream>>>(deg, esum, lattr);

  // x -> f16
  k_cvt_f16<<<(NN * DIN + T - 1) / T, T, 0, stream>>>(x, x16, NN * DIN);

  const half_t* Ain = x16;
  int K = DIN;
  for (int l = 0; l < 3; ++l) {
    k_cvt_wT<<<(K * HCC + T - 1) / T, T, 0, stream>>>(W[l], WT16, K);
    k_coef<<<1, 64, 0, stream>>>(We[l], Ae[l], coef);

    // projection GEMM: 625 M-strips x 4 N-super-tiles, 8 waves per block
    k_gemm_wmma<<<625 * (HCC / 128), T, 0, stream>>>(Ain, WT16, hproj, K);

    k_scores<<<(NN * HH + T - 1) / T, T, 0, stream>>>(hproj, As[l], Ad[l], ssrc, sdst);

    (void)hipMemsetAsync(amax,  0, (size_t)NN * HH * 4, stream);   // key 0 < key(-inf)
    (void)hipMemsetAsync(denom, 0, (size_t)NN * HH * 4, stream);
    (void)hipMemsetAsync(hagg,  0, (size_t)NN * HCC * 4, stream);

    k_edge_alpha<<<(ET + T - 1) / T, T, 0, stream>>>(ei, eattr, lattr, ssrc, sdst,
                                                     coef, alpha, amax);
    k_edge_ex<<<(ET + T - 1) / T, T, 0, stream>>>(ei, alpha, amax, denom);
    long long aggT = (long long)ET * (HH * 16);
    k_edge_agg<<<(int)((aggT + T - 1) / T), T, 0, stream>>>(ei, alpha, denom, hproj, hagg);

    k_finalize<<<(NN * HCC + T - 1) / T, T, 0, stream>>>(hagg, B[l], h16);
    Ain = h16; K = HCC;
  }

  // head + pooling
  (void)hipMemsetAsync(pooled, 0, (size_t)GG * ACTD * 4, stream);
  (void)hipMemsetAsync(cnt,    0, (size_t)GG * 4, stream);
  k_cnt<<<(NN + T - 1) / T, T, 0, stream>>>(batch, cnt);
  k_head<<<(NN * ACTD + T - 1) / T, T, 0, stream>>>(h16, Wl, bl, batch, pooled);
  k_pool_div<<<(GG * ACTD + T - 1) / T, T, 0, stream>>>(pooled, cnt);
}
